// net_gcn_8315056685239
// MI455X (gfx1250) — matrix operand order repaired and checked by reading, not compiled, not run on hardware
//
#include <hip/hip_runtime.h>
#include <hip/hip_bf16.h>
#include <stdint.h>

#define N_NODES 8192
#define F0D 512
#define F1D 256
#define F2D 16

typedef __bf16 v16bf __attribute__((ext_vector_type(16)));
typedef float  v8f   __attribute__((ext_vector_type(8)));

union FragBF { v16bf v; unsigned short u[16]; };

__device__ __forceinline__ unsigned short f2bf(float f) {
  // round-to-nearest-even f32 -> bf16
  unsigned int u = __float_as_uint(f);
  unsigned int r = u + 0x7FFFu + ((u >> 16) & 1u);
  return (unsigned short)(r >> 16);
}
__device__ __forceinline__ float bf2f(unsigned short u) {
  return __uint_as_float((unsigned int)u << 16);
}

// CDNA5 async DMA: global -> LDS, 16B per lane, tracked by ASYNCcnt.
// LDS byte address comes from the vdst VGPR; GV addressing (64-bit vaddr, saddr=off).
__device__ __forceinline__ void async_b128(unsigned ldsByteOff, const void* gptr) {
  asm volatile("global_load_async_to_lds_b128 %0, %1, off"
               :: "v"(ldsByteOff), "v"(gptr) : "memory");
}
__device__ __forceinline__ void wait_async0() {
  asm volatile("s_wait_asynccnt 0x0" ::: "memory");
}

// ---------------- pass 1: rowsum -> dinv, and cache adj as bf16 ----------------
__global__ __launch_bounds__(256) void gcn_cvtrow(const float* __restrict__ adj,
                                                  unsigned short* __restrict__ adjb,
                                                  float* __restrict__ dinv, int n) {
  __shared__ float red[256];
  const int row = blockIdx.x;
  const float* a = adj + (size_t)row * n;
  unsigned short* o = adjb + (size_t)row * n;
  float s = 0.f;
  for (int j = threadIdx.x * 4; j < n; j += 1024) {
    float4 v = *(const float4*)(a + j);
    s += v.x + v.y + v.z + v.w;
    uint2 p;
    p.x = (unsigned)f2bf(v.x) | ((unsigned)f2bf(v.y) << 16);
    p.y = (unsigned)f2bf(v.z) | ((unsigned)f2bf(v.w) << 16);
    *(uint2*)(o + j) = p;
  }
  red[threadIdx.x] = s; __syncthreads();
  for (int off = 128; off > 0; off >>= 1) {
    if (threadIdx.x < off) red[threadIdx.x] += red[threadIdx.x + off];
    __syncthreads();
  }
  if (threadIdx.x == 0) {
    float deg = red[0] + 1.0f;             // +I self loop
    dinv[row] = (deg > 0.f) ? rsqrtf(deg) : 0.f;
  }
}

__global__ __launch_bounds__(256) void gcn_roww(const unsigned short* __restrict__ adjb,
                                                const float* __restrict__ dinv,
                                                float* __restrict__ roww, int n) {
  __shared__ float red[256];
  const int row = blockIdx.x;
  const unsigned short* a = adjb + (size_t)row * n;
  float s = 0.f;
  for (int j = threadIdx.x * 4; j < n; j += 1024) {
    uint2 q = *(const uint2*)(a + j);
    s += bf2f((unsigned short)(q.x & 0xffff)) * dinv[j + 0];
    s += bf2f((unsigned short)(q.x >> 16))    * dinv[j + 1];
    s += bf2f((unsigned short)(q.y & 0xffff)) * dinv[j + 2];
    s += bf2f((unsigned short)(q.y >> 16))    * dinv[j + 3];
  }
  red[threadIdx.x] = s; __syncthreads();
  for (int off = 128; off > 0; off >>= 1) {
    if (threadIdx.x < off) red[threadIdx.x] += red[threadIdx.x + off];
    __syncthreads();
  }
  if (threadIdx.x == 0) roww[row] = red[0];
}

__global__ __launch_bounds__(256) void gcn_colw(const unsigned short* __restrict__ adjb,
                                                const float* __restrict__ dinv,
                                                float* __restrict__ colw, int n) {
  const int col = blockIdx.x * 256 + threadIdx.x;
  float s = 0.f;
  for (int i = 0; i < n; ++i) s += bf2f(adjb[(size_t)i * n + col]) * dinv[i];
  colw[col] = s;
}

__global__ __launch_bounds__(256) void gcn_rs(const float* __restrict__ dinv,
                                              const float* __restrict__ roww,
                                              const float* __restrict__ colw,
                                              float* __restrict__ rs,
                                              float* __restrict__ rinv, int n) {
  const int i = blockIdx.x * 256 + threadIdx.x;
  if (i >= n) return;
  float di = dinv[i];
  float r = 0.5f * di * (roww[i] + colw[i] + 2.0f * di);
  rs[i] = r;
  float t = r + 0.001f;
  rinv[i] = (t > 0.f) ? rsqrtf(t) : 0.f;
}

// ---------------- B operands, K-major (column-major) bf16 ----------------
// xcm[c*8192 + i]: c<512 -> dinv[i]*x[i,c] (U) ; c>=512 -> dinv[i]*rinv[i]*x[i,c-512] (Z)
__global__ __launch_bounds__(256) void gcn_xcat(const float* __restrict__ x,
                                                const float* __restrict__ dinv,
                                                const float* __restrict__ rinv,
                                                unsigned short* __restrict__ xcm) {
  const size_t idx = (size_t)blockIdx.x * 256 + threadIdx.x;  // = c*8192 + i
  const int c = (int)(idx >> 13);
  const int i = (int)(idx & 8191);
  float xv = x[(size_t)i * F0D + (c & 511)];
  float sc = dinv[i] * ((c < 512) ? 1.0f : rinv[i]);
  xcm[idx] = f2bf(xv * sc);
}

// w0cm[n*512 + k] = bf16(W0[k,n])
__global__ __launch_bounds__(256) void gcn_cvtw0(const float* __restrict__ W0,
                                                 unsigned short* __restrict__ w0cm) {
  const int idx = blockIdx.x * 256 + threadIdx.x;   // = n*512 + k
  const int n = idx >> 9, k = idx & 511;
  w0cm[idx] = f2bf(W0[(size_t)k * F1D + n]);
}

// ---------------- WMMA bf16 GEMM: C[M,ncols] = A[M,K] @ B[K,ncols] ----------------
// A: bf16 row-major (lda=K). B: bf16 K-major (column n at B + n*K).
// Block tile 128x256, 8 waves (4 row x 2 col), wave tile 32x128 = 2x8 frags.
// Double-buffered LDS tiles filled by global_load_async_to_lds_b128 (ASYNCcnt).
// MODE 0: store f32 row-major.  MODE 1: relu -> Cf f32 row-major, Cbf = bf16(dinv[row]*relu)
//         stored K-major at Cbf[col*N_NODES + row] (feeds next GEMM as B).
template <int MODE>
__global__ __launch_bounds__(256) void gemm_bf16_k(const unsigned short* __restrict__ A,
                                                   const unsigned short* __restrict__ B,
                                                   float* __restrict__ Cf,
                                                   unsigned short* __restrict__ Cbf,
                                                   const float* __restrict__ dinv,
                                                   int K, int ncols) {
  __shared__ __align__(16) unsigned short As[2 * 128 * 32];  // [buf][r][k]
  __shared__ __align__(16) unsigned short Bs[2 * 256 * 32];  // [buf][n][k]

  const int tid  = threadIdx.x;
  const int lane = tid & 31, wid = tid >> 5;
  const int wr = wid & 3, wc = wid >> 2;        // 4x2 wave grid
  const int rb = wr * 32, cb = wc * 128;
  const int lh = lane & 15;
  const bool hihalf = lane >= 16;
  const int m0 = blockIdx.y * 128;
  const int n0 = blockIdx.x * 256;

  const unsigned asBase = (unsigned)(size_t)(void*)As;
  const unsigned bsBase = (unsigned)(size_t)(void*)Bs;

  // staging coords: A = 128x32 (32B/thread), B = 256x32 (64B/thread, one column)
  const int ra = tid >> 1, ca = (tid & 1) << 4;

  v8f acc[2][8];
#pragma unroll
  for (int s = 0; s < 2; ++s)
#pragma unroll
    for (int c = 0; c < 8; ++c) acc[s][c] = (v8f){0, 0, 0, 0, 0, 0, 0, 0};

  auto stage = [&](int k0, int buf) {
    const unsigned short* sa = A + (size_t)(m0 + ra) * (size_t)K + k0 + ca;
    unsigned da = asBase + (unsigned)((buf * 4096 + ra * 32 + ca) * 2);
    async_b128(da,      sa);
    async_b128(da + 16, sa + 8);
    const unsigned short* sb = B + (size_t)(n0 + tid) * (size_t)K + k0;
    unsigned db = bsBase + (unsigned)((buf * 8192 + tid * 32) * 2);
    async_b128(db,      sb);
    async_b128(db + 16, sb + 8);
    async_b128(db + 32, sb + 16);
    async_b128(db + 48, sb + 24);
  };

  stage(0, 0);
  wait_async0();
  __syncthreads();

  int buf = 0;
  for (int k0 = 0; k0 < K; k0 += 32) {
    if (k0 + 32 < K) stage(k0 + 32, buf ^ 1);   // DMA next tile while computing

    const unsigned short* AsBuf = &As[buf * 4096];
    const unsigned short* BsBuf = &Bs[buf * 8192];

    // A fragments (contiguous layout: lanes 16-31 hold K=16..31)
    FragBF a[2];
    const int kbA = hihalf ? 16 : 0;
#pragma unroll
    for (int s = 0; s < 2; ++s) {
      const unsigned short* base = &AsBuf[(rb + s * 16 + lh) * 32];
#pragma unroll
      for (int j = 0; j < 8; ++j) {
        a[s].u[j]     = base[kbA + j];
        a[s].u[8 + j] = base[kbA + 8 + j];
      }
    }
    const int kbB = hihalf ? 16 : 0;            // B: lanes 16-31 hold K=16..31
#pragma unroll
    for (int c = 0; c < 8; ++c) {
      FragBF b;
      const unsigned short* bb = &BsBuf[(cb + c * 16 + lh) * 32 + kbB];
#pragma unroll
      for (int j = 0; j < 16; ++j) b.u[j] = bb[j];
      acc[0][c] = __builtin_amdgcn_wmma_f32_16x16x32_bf16(
          false, a[0].v, false, b.v, (short)0, acc[0][c], false, false);
      acc[1][c] = __builtin_amdgcn_wmma_f32_16x16x32_bf16(
          false, a[1].v, false, b.v, (short)0, acc[1][c], false, false);
    }
    wait_async0();
    __syncthreads();
    buf ^= 1;
  }

  // C layout: VGPR v, lane L -> row = base + v + (L>=16 ? 8 : 0), col = base + L%16
#pragma unroll
  for (int s = 0; s < 2; ++s) {
#pragma unroll
    for (int c = 0; c < 8; ++c) {
      const int col = n0 + cb + c * 16 + lh;
#pragma unroll
      for (int v = 0; v < 8; ++v) {
        const int row = m0 + rb + s * 16 + v + (hihalf ? 8 : 0);
        const float val = acc[s][c][v];
        if (MODE == 0) {
          Cf[(size_t)row * (size_t)ncols + col] = val;
        } else {
          float h = val > 0.f ? val : 0.f;                       // relu
          Cf[(size_t)row * (size_t)ncols + col] = h;             // h2 (f32)
          Cbf[(size_t)col * N_NODES + row] = f2bf(dinv[row] * h); // V, K-major
        }
      }
    }
  }
}

// ---------------- epilogue-1: h1 (bf16) + fused loss partials ----------------
// P[:, :512] = adj@U ; P[:, 512:] = adj@Z
// h1 = dinv*A1 + dinv^2*x
// loss_elem = rinv^2*rs*x^2 - ( Z*A2 + dinv^2*rinv^2*x^2 ),  Z = dinv*rinv*x
__global__ __launch_bounds__(256) void gcn_epi1(const float* __restrict__ P,
                                                const float* __restrict__ x,
                                                const float* __restrict__ dinv,
                                                const float* __restrict__ rinv,
                                                const float* __restrict__ rs,
                                                unsigned short* __restrict__ h1b,
                                                float* __restrict__ partials) {
  __shared__ float red[256];
  const size_t idx = (size_t)blockIdx.x * 256 + threadIdx.x;
  const int i = (int)(idx >> 9);
  const int k = (int)(idx & 511);
  const float a1 = P[(size_t)i * 1024 + k];
  const float a2 = P[(size_t)i * 1024 + 512 + k];
  const float xv = x[idx];
  const float di = dinv[i], ri = rinv[i];
  h1b[idx] = f2bf(di * a1 + di * di * xv);
  const float Z = di * ri * xv;
  const float contrib = ri * ri * rs[i] * xv * xv - (Z * a2 + di * di * ri * ri * xv * xv);
  red[threadIdx.x] = contrib; __syncthreads();
  for (int o = 128; o > 0; o >>= 1) {
    if (threadIdx.x < o) red[threadIdx.x] += red[threadIdx.x + o];
    __syncthreads();
  }
  if (threadIdx.x == 0) partials[blockIdx.x] = red[0];
}

__global__ __launch_bounds__(256) void gcn_loss_reduce(const float* __restrict__ partials,
                                                       int n, float* __restrict__ outp) {
  __shared__ float red[256];
  float s = 0.f;
  for (int i = threadIdx.x; i < n; i += 256) s += partials[i];
  red[threadIdx.x] = s; __syncthreads();
  for (int o = 128; o > 0; o >>= 1) {
    if (threadIdx.x < o) red[threadIdx.x] += red[threadIdx.x + o];
    __syncthreads();
  }
  if (threadIdx.x == 0) outp[0] = red[0];
}

// ---------------- final: h3 = dinv*B2 + dinv^2*h2 ; out = h3 @ W1 ----------------
__global__ __launch_bounds__(256) void gcn_final(const float* __restrict__ B2,
                                                 const float* __restrict__ h2f,
                                                 const float* __restrict__ dinv,
                                                 const float* __restrict__ W1,
                                                 float* __restrict__ out) {
  __shared__ float w1s[F1D * F2D];
  for (int t = threadIdx.x; t < F1D * F2D; t += 256) w1s[t] = W1[t];
  __syncthreads();
  const int i = blockIdx.x * 256 + threadIdx.x;
  const float di = dinv[i], di2 = di * di;
  float acc[F2D];
#pragma unroll
  for (int c = 0; c < F2D; ++c) acc[c] = 0.f;
  for (int k = 0; k < F1D; ++k) {
    const float h3 = di * B2[(size_t)i * F1D + k] + di2 * h2f[(size_t)i * F1D + k];
    const float* wv = &w1s[k * F2D];
#pragma unroll
    for (int c = 0; c < F2D; ++c) acc[c] += h3 * wv[c];
  }
#pragma unroll
  for (int c = 0; c < F2D; ++c) out[(size_t)i * F2D + c] = acc[c];
}

// ---------------- launch ----------------
extern "C" void kernel_launch(void* const* d_in, const int* in_sizes, int n_in,
                              void* d_out, int out_size, void* d_ws, size_t ws_size,
                              hipStream_t stream) {
  const float* x   = (const float*)d_in[0];
  const float* adj = (const float*)d_in[1];
  // d_in[2], d_in[3] (mask1, mask2) unused: mask = (adj!=0) -> adj*mask1*mask2 == adj identically
  const float* W0  = (const float*)d_in[4];
  const float* W1  = (const float*)d_in[5];
  float* out = (float*)d_out;

  const int N = N_NODES;
  char* w = (char*)d_ws;
  size_t off = 0;
  auto alloc = [&](size_t bytes) -> char* {
    char* p = w + off;
    off = (off + bytes + 255) & ~(size_t)255;
    return p;
  };
  float* dinv          = (float*)alloc((size_t)N * 4);
  float* roww          = (float*)alloc((size_t)N * 4);
  float* colw          = (float*)alloc((size_t)N * 4);
  float* rs            = (float*)alloc((size_t)N * 4);
  float* rinv          = (float*)alloc((size_t)N * 4);
  float* partials      = (float*)alloc((size_t)(N * F0D / 256) * 4);     // 16384
  unsigned short* adjb = (unsigned short*)alloc((size_t)N * N * 2);      // adj bf16 cache
  unsigned short* xcm  = (unsigned short*)alloc((size_t)N * 1024 * 2);   // [U|Z] K-major
  float* P             = (float*)alloc((size_t)N * 1024 * 4);            // adj@[U|Z]
  unsigned short* h1b  = (unsigned short*)alloc((size_t)N * F0D * 2);    // h1 row-major
  unsigned short* w0cm = (unsigned short*)alloc((size_t)F0D * F1D * 2);  // W0 K-major
  float* h2f           = (float*)alloc((size_t)N * F1D * 4);
  unsigned short* vcm  = (unsigned short*)alloc((size_t)N * F1D * 2);    // V K-major
  float* b2            = (float*)alloc((size_t)N * F1D * 4);

  gcn_cvtrow<<<N, 256, 0, stream>>>(adj, adjb, dinv, N);
  gcn_roww<<<N, 256, 0, stream>>>(adjb, dinv, roww, N);
  gcn_colw<<<N / 256, 256, 0, stream>>>(adjb, dinv, colw, N);
  gcn_rs<<<N / 256, 256, 0, stream>>>(dinv, roww, colw, rs, rinv, N);
  gcn_xcat<<<(N * 1024) / 256, 256, 0, stream>>>(x, dinv, rinv, xcm);
  gcn_cvtw0<<<(F0D * F1D) / 256, 256, 0, stream>>>(W0, w0cm);

  // P = adj @ [U|Z]  (8192x8192x1024)
  gemm_bf16_k<0><<<dim3(1024 / 256, N / 128), 256, 0, stream>>>(
      adjb, xcm, P, nullptr, nullptr, N, 1024);

  gcn_epi1<<<(N * F0D) / 256, 256, 0, stream>>>(P, x, dinv, rinv, rs, h1b, partials);
  gcn_loss_reduce<<<1, 256, 0, stream>>>(partials, N * F0D / 256, out + (size_t)N * F2D);

  // h2 = relu(h1 @ W0); V = dinv ⊙ h2 (bf16, K-major)
  gemm_bf16_k<1><<<dim3(1, N / 128), 256, 0, stream>>>(
      h1b, w0cm, h2f, vcm, dinv, F0D, F1D);

  // B2 = adj @ V  (8192x8192x256)
  gemm_bf16_k<0><<<dim3(1, N / 128), 256, 0, stream>>>(
      adjb, vcm, b2, nullptr, nullptr, N, F1D);

  // out = (dinv⊙B2 + dinv²⊙h2) @ W1
  gcn_final<<<N / 256, 256, 0, stream>>>(b2, h2f, dinv, W1, out);
}
